// EquiLocalPatOrientConvolution_31190052504468
// MI455X (gfx1250) — matrix operand order repaired and checked by reading, not compiled, not run on hardware
//
#include <hip/hip_runtime.h>
#include <cstdint>

// CDNA5 / gfx1250 implicit-GEMM 3D conv via v_wmma_f32_16x16x32_f16.
// GEMM: M = 64 out-channels, N = spatial, K = 125 taps x 32 in-channels.
// Round 3: 2x3 register tile per wave (1.67 mem ops per WMMA); each n-group
// owns exactly one 48-wide output row -> no div/mod in B addressing.

typedef __attribute__((ext_vector_type(16))) _Float16 v16h;
typedef __attribute__((ext_vector_type(8)))  float    v8f;

#define Y0C   0.28209479177387814f
#define O_CH  64
#define I_CH  32
#define PCNT  10
#define DHW   48
#define TAPS  125   // 5*5*5

// ---------------------------------------------------------------------------
// Filter build + WMMA A-fragment swizzle.
// fa dword index = ((tap*4 + mtile)*32 + lane)*8 + d   (packed f16x2)
// A 16x32 f16 layout (ISA 7.12.2): lanes 0-15: M=lane, dwords hold K pairs
// {0,1}{2,3}{4,5}{6,7}{16,17}{18,19}{20,21}{22,23}; lanes 16-31: K base +8.
// K within a tile == input channel i (tile == tap).
// ---------------------------------------------------------------------------
__global__ void build_filter_kernel(const float* __restrict__ W,      // [64][32][10]
                                    const float* __restrict__ masks,  // [10][125]
                                    uint32_t* __restrict__ fa)
{
  int idx = blockIdx.x * blockDim.x + threadIdx.x;
  if (idx >= TAPS * 4 * 32 * 8) return;
  int d     = idx & 7;
  int lane  = (idx >> 3) & 31;
  int mtile = (idx >> 8) & 3;
  int tap   = idx >> 10;
  int m = mtile * 16 + (lane & 15);
  int k = ((d < 4) ? 2 * d : 8 + 2 * d) + ((lane & 16) ? 8 : 0);
  float f0 = 0.f, f1 = 0.f;
#pragma unroll
  for (int p = 0; p < PCNT; ++p) {
    float mk = masks[p * TAPS + tap];
    f0 += W[(m * I_CH + k    ) * PCNT + p] * mk;
    f1 += W[(m * I_CH + k + 1) * PCNT + p] * mk;
  }
  union { _Float16 h[2]; uint32_t u; } pk;
  pk.h[0] = (_Float16)(f0 * Y0C);
  pk.h[1] = (_Float16)(f1 * Y0C);
  fa[idx] = pk.u;
}

// ---------------------------------------------------------------------------
// Main conv kernel. Block = (b, z, y-quad): 192 positions x 64 channels.
// 8 waves (256 thr): mg = wave&1 (mtiles 2mg,2mg+1), ng = wave>>1 (0..3) owns
// output row y0+ng, columns ng*48 + {0..47} as three 16-col N-tiles.
// Each wave: 2x3 accumulator tile -> 6 WMMAs per (2 A + 3 B) fragment loads.
// LDS slab (one dz at a time): [yy 0..7][xx 0..51][ch 0..31] f16, channel-
// innermost so a B fragment is two contiguous ds_load_b128 per lane.
// ---------------------------------------------------------------------------
__global__ __launch_bounds__(256)
void conv_wmma_kernel(const float* __restrict__ x,     // [2][32][48][48][48]
                      const float* __restrict__ bias,  // [64]
                      const uint32_t* __restrict__ fa, // swizzled filter
                      float* __restrict__ out)         // [2][64][48][48][48]
{
  __shared__ __align__(16) _Float16 patch[8 * 52 * 32]; // 26624 B

  const int tid  = threadIdx.x;
  const int lane = tid & 31;
  const int wave = tid >> 5;
  const int mg   = wave & 1;     // m-group: mtiles 2mg, 2mg+1
  const int ng   = wave >> 1;    // n-group: output row y0+ng, cols ng*48+..

  int bid = blockIdx.x;
  const int yq = bid % 12;  bid /= 12;
  const int z  = bid % DHW;
  const int b  = bid / DHW;
  const int y0 = yq * 4;

  // Three N-tiles of this wave share one patch row (yl == ng):
  // columns xp0 = lane&15, xp0+16, xp0+32 within row y0+ng.
  const int xp0 = lane & 15;
  const int chb = (lane & 16) >> 1;           // B: lanes 16-31 -> K 8..15, 24..31
  const int bB  = (ng * 52 + xp0) * 32 + chb; // f16 index, N-tile stride 16*32

  v8f acc00 = {}, acc01 = {}, acc02 = {};
  v8f acc10 = {}, acc11 = {}, acc12 = {};

  for (int dz = 0; dz < 5; ++dz) {
    __syncthreads();
    // ---- stage z-slab (z-2+dz): 8 y-rows x 52 x x 32 ch, f32 -> f16 ----
    const int  zz  = z - 2 + dz;
    const bool zok = (zz >= 0) && (zz < DHW);
    for (int e = tid; e < 8 * 52 * 32; e += 256) {
      int xx = e % 52;                        // x innermost -> coalesced
      int r  = e / 52;
      int ch = r & 31;
      int yy = r >> 5;                        // 0..7
      int yg = y0 - 2 + yy;
      int xg = xx - 2;
      float v = 0.f;
      if (zok && (unsigned)yg < DHW && (unsigned)xg < DHW)
        v = x[(((size_t)(b * I_CH + ch) * DHW + zz) * DHW + yg) * DHW + xg];
      patch[(yy * 52 + xx) * 32 + ch] = (_Float16)v;
    }
    __syncthreads();

    // ---- 25 taps of this slab: K-tile per tap, 6 WMMAs per tap ----
#pragma unroll
    for (int t = 0; t < 25; ++t) {
      const int dy = t / 5, dx = t % 5;

      // A fragments for mtiles 2mg, 2mg+1: global_load_b128 x4 (L2-hot).
      const size_t abase = (size_t)((((dz * 25 + t) * 4 + 2 * mg) * 32 + lane) * 8);
      const uint4* ap0 = reinterpret_cast<const uint4*>(fa + abase);
      const uint4* ap1 = reinterpret_cast<const uint4*>(fa + abase + 32 * 8);
      v16h a0, a1;
      ((uint4*)&a0)[0] = ap0[0];
      ((uint4*)&a0)[1] = ap0[1];
      ((uint4*)&a1)[0] = ap1[0];
      ((uint4*)&a1)[1] = ap1[1];

      // B fragments: two ds_load_b128 each; N-tiles offset by 16 cols = 512 h.
      const int off = bB + (dy * 52 + dx) * 32;
      v16h b0, b1, b2;
      ((uint4*)&b0)[0] = *reinterpret_cast<const uint4*>(&patch[off]);
      ((uint4*)&b0)[1] = *reinterpret_cast<const uint4*>(&patch[off + 16]);
      ((uint4*)&b1)[0] = *reinterpret_cast<const uint4*>(&patch[off + 16 * 32]);
      ((uint4*)&b1)[1] = *reinterpret_cast<const uint4*>(&patch[off + 16 * 32 + 16]);
      ((uint4*)&b2)[0] = *reinterpret_cast<const uint4*>(&patch[off + 32 * 32]);
      ((uint4*)&b2)[1] = *reinterpret_cast<const uint4*>(&patch[off + 32 * 32 + 16]);

      acc00 = __builtin_amdgcn_wmma_f32_16x16x32_f16(
          false, a0, false, b0, (short)0, acc00, false, false);
      acc01 = __builtin_amdgcn_wmma_f32_16x16x32_f16(
          false, a0, false, b1, (short)0, acc01, false, false);
      acc02 = __builtin_amdgcn_wmma_f32_16x16x32_f16(
          false, a0, false, b2, (short)0, acc02, false, false);
      acc10 = __builtin_amdgcn_wmma_f32_16x16x32_f16(
          false, a1, false, b0, (short)0, acc10, false, false);
      acc11 = __builtin_amdgcn_wmma_f32_16x16x32_f16(
          false, a1, false, b1, (short)0, acc11, false, false);
      acc12 = __builtin_amdgcn_wmma_f32_16x16x32_f16(
          false, a1, false, b2, (short)0, acc12, false, false);
    }
  }

  // ---- epilogue: C/D layout -> lane = N, VGPR v = M (+8 for upper half) ----
  const int mrow = (lane & 16) ? 8 : 0;
  const size_t rowb = (size_t)z * (DHW * DHW) + (size_t)(y0 + ng) * DHW + xp0;
#pragma unroll
  for (int v = 0; v < 8; ++v) {
    const int o0 = (2 * mg    ) * 16 + mrow + v;
    const int o1 = (2 * mg + 1) * 16 + mrow + v;
    const float bv0 = bias[o0];
    const float bv1 = bias[o1];
    float* p0 = out + (size_t)(b * O_CH + o0) * (DHW * DHW * DHW) + rowb;
    float* p1 = out + (size_t)(b * O_CH + o1) * (DHW * DHW * DHW) + rowb;
    p0[0]  = acc00[v] + bv0;
    p0[16] = acc01[v] + bv0;
    p0[32] = acc02[v] + bv0;
    p1[0]  = acc10[v] + bv1;
    p1[16] = acc11[v] + bv1;
    p1[32] = acc12[v] + bv1;
  }
}

// ---------------------------------------------------------------------------
extern "C" void kernel_launch(void* const* d_in, const int* in_sizes, int n_in,
                              void* d_out, int out_size, void* d_ws, size_t ws_size,
                              hipStream_t stream) {
  const float* x     = (const float*)d_in[0];  // [2,32,1,48,48,48]
  const float* w     = (const float*)d_in[1];  // [64,32,1,1,1,10]
  const float* bias  = (const float*)d_in[2];  // [64]
  const float* masks = (const float*)d_in[3];  // [10,5,5,5]
  float*       out   = (float*)d_out;          // [2,64,1,48,48,48]
  uint32_t*    fa    = (uint32_t*)d_ws;        // needs 512000 B of scratch

  // Build + swizzle filter into WMMA A-fragment layout.
  const int fa_elems = TAPS * 4 * 32 * 8;      // 128000 dwords
  build_filter_kernel<<<(fa_elems + 255) / 256, 256, 0, stream>>>(w, masks, fa);

  // Implicit-GEMM convolution: 2*48*12 = 1152 blocks of 256 threads.
  conv_wmma_kernel<<<2 * DHW * (DHW / 4), 256, 0, stream>>>(x, bias, fa, out);
}
